// GNNModel_7937099563012
// MI455X (gfx1250) — matrix operand order repaired and checked by reading, not compiled, not run on hardware
//
#include <hip/hip_runtime.h>

// ---------------------------------------------------------------------------
// SR-GNN forward for MI455X (gfx1250, wave32, WMMA).
// All GEMMs run through one wave-tiled, software-pipelined
// v_wmma_f32_16x16x32_f16 kernel (K, dual-A, activation, outputs templated).
// ---------------------------------------------------------------------------

typedef __attribute__((ext_vector_type(16))) _Float16 v16h;
typedef __attribute__((ext_vector_type(8)))  _Float16 v8h;
typedef __attribute__((ext_vector_type(8)))  float    v8f;

#define HDIM   128
#define SESS   1024
#define NPER   50
#define NROWS  (SESS * NPER)     // 51200
#define NEDGE  (8 * NROWS)       // 409600
#define NNODE  100000

// ---- fast transcendentals (single HW ops on gfx1250) ----------------------
__device__ inline float fast_rcp(float x) { return __builtin_amdgcn_rcpf(x); }

__device__ inline float fast_sigmoid(float x) {
    return fast_rcp(1.0f + __expf(-x));           // v_exp_f32 + v_rcp_f32
}

__device__ inline float fast_tanh(float x) {
#if __has_builtin(__builtin_amdgcn_tanhf)
    return __builtin_amdgcn_tanhf(x);             // v_tanh_f32
#else
    return 1.0f - 2.0f * fast_rcp(__expf(2.0f * x) + 1.0f);
#endif
}

// ---- WMMA fragment loaders (contiguous k order, same for A and B) ---------------------------
// A (16x32 f16, M across lanes 0..15 twice): lane<16 holds K k0..k0+15,
// lane>=16 holds K k0+16..k0+31. Two 16-byte loads.
__device__ inline v16h load_a_frag(const _Float16* __restrict__ A, int lda,
                                   int row, int k0, int lane) {
    const int off8 = (lane >= 16) ? 16 : 0;
    const _Float16* p = A + (size_t)row * lda + k0 + off8;
    v8h lo = *(const v8h*)p;
    v8h hi = *(const v8h*)(p + 8);
    v16h a;
#pragma unroll
    for (int i = 0; i < 8; ++i) { a[i] = lo[i]; a[i + 8] = hi[i]; }
    return a;
}

// B stored pre-transposed as Bt[Nout, K] row-major. Lane l holds column n0+(l&15);
// lanes 0..15 K = k0..k0+15, lanes 16..31 K = k0+16..k0+31. One 32-byte load.
__device__ inline v16h load_b_frag(const _Float16* __restrict__ Bt, int ldb,
                                   int col, int k0, int lane) {
    const int off16 = (lane >= 16) ? 16 : 0;
    const _Float16* p = Bt + (size_t)col * ldb + k0 + off16;
    return *(const v16h*)p;
}

// ---- Generic WMMA GEMM ----------------------------------------------------
// C[M,Nout] = act( A[M,K] @ Bt^T + bias + bias2  [+ Avn@B2t^T when DUAL] )
// DUAL: fused attention path, second accumulation with A rows remapped to the
//       session's last node ( (r/50)*50+49 ).
// MREP > 1: one wave computes MREP vertical tiles reusing each B fragment.
// NK = K/32. K loop fully unrolled + double-buffered so the next k-step's
// global_load_b128s are in flight during the current v_wmma.
// ACT: 0 = identity, 1 = sigmoid.  O32/O16: which outputs to store.
template <int MREP, int NK, bool DUAL, int ACT, bool O32, bool O16>
__global__ void k_gemm(const _Float16* __restrict__ A, int lda,
                       const _Float16* __restrict__ Bt,
                       const _Float16* __restrict__ B2t, int ldb,
                       const float* __restrict__ bias,
                       const float* __restrict__ bias2,
                       float* __restrict__ C32, _Float16* __restrict__ C16,
                       int ldc, int M, int Nout) {
    const int lane = threadIdx.x & 31;
    const int wave = blockIdx.x * (blockDim.x >> 5) + (threadIdx.x >> 5);
    const int ntN = Nout >> 4;
    const int ntiles = (M / (16 * MREP)) * ntN;
    if (wave >= ntiles) return;                 // wave-uniform: EXEC stays all-1s
    const int m0 = (wave / ntN) * (16 * MREP);
    const int n0 = (wave % ntN) * 16;
    const int l15 = lane & 15;
    const int bcol = n0 + l15;

    int arow[MREP], vrow[MREP];
#pragma unroll
    for (int t = 0; t < MREP; ++t) {
        arow[t] = m0 + t * 16 + l15;
        if (DUAL) vrow[t] = (arow[t] / NPER) * NPER + (NPER - 1);   // v_n[batch[r]]
    }

    v8f acc[MREP] = {};

    // -------- software pipeline: prime k-step 0 --------
    v16h bcur = load_b_frag(Bt, ldb, bcol, 0, lane);
    v16h b2cur, a2cur[MREP];
    v16h acur[MREP];
#pragma unroll
    for (int t = 0; t < MREP; ++t) acur[t] = load_a_frag(A, lda, arow[t], 0, lane);
    if (DUAL) {
        b2cur = load_b_frag(B2t, ldb, bcol, 0, lane);
#pragma unroll
        for (int t = 0; t < MREP; ++t) a2cur[t] = load_a_frag(A, lda, vrow[t], 0, lane);
    }

#pragma unroll
    for (int ks = 0; ks < NK; ++ks) {
        v16h bnxt, b2nxt, anxt[MREP], a2nxt[MREP];
        if (ks + 1 < NK) {                       // issue next k-step's loads first
            const int kn = (ks + 1) * 32;
            bnxt = load_b_frag(Bt, ldb, bcol, kn, lane);
#pragma unroll
            for (int t = 0; t < MREP; ++t) anxt[t] = load_a_frag(A, lda, arow[t], kn, lane);
            if (DUAL) {
                b2nxt = load_b_frag(B2t, ldb, bcol, kn, lane);
#pragma unroll
                for (int t = 0; t < MREP; ++t) a2nxt[t] = load_a_frag(A, lda, vrow[t], kn, lane);
            }
        }
#pragma unroll
        for (int t = 0; t < MREP; ++t)
            acc[t] = __builtin_amdgcn_wmma_f32_16x16x32_f16(
                false, acur[t], false, bcur, (short)0, acc[t], false, false);
        if (DUAL) {
#pragma unroll
            for (int t = 0; t < MREP; ++t)
                acc[t] = __builtin_amdgcn_wmma_f32_16x16x32_f16(
                    false, a2cur[t], false, b2cur, (short)0, acc[t], false, false);
        }
        if (ks + 1 < NK) {
            bcur = bnxt;
#pragma unroll
            for (int t = 0; t < MREP; ++t) acur[t] = anxt[t];
            if (DUAL) {
                b2cur = b2nxt;
#pragma unroll
                for (int t = 0; t < MREP; ++t) a2cur[t] = a2nxt[t];
            }
        }
    }

    // -------- epilogue: bias + optional sigmoid, f32/f16 stores --------
    float badd = 0.0f;
    if (bias)  badd += bias[bcol];
    if (bias2) badd += bias2[bcol];
    const int hi8 = (lane >= 16) ? 8 : 0;
#pragma unroll
    for (int t = 0; t < MREP; ++t) {
#pragma unroll
        for (int r = 0; r < 8; ++r) {
            float v = acc[t][r] + badd;
            if (ACT == 1) v = fast_sigmoid(v);
            const size_t idx = (size_t)(m0 + t * 16 + hi8 + r) * (size_t)ldc + bcol;
            if (O32) C32[idx] = v;
            if (O16) C16[idx] = (_Float16)v;
        }
    }
}

// ---- Small helper kernels -------------------------------------------------
__global__ void k_f32_to_f16(const float* __restrict__ in,
                             _Float16* __restrict__ out, int n) {
    int i = blockIdx.x * blockDim.x + threadIdx.x;
    if (i < n) out[i] = (_Float16)in[i];
}

__global__ void k_zero_f32(float* __restrict__ p, int n) {
    int i = blockIdx.x * blockDim.x + threadIdx.x;
    if (i < n) p[i] = 0.0f;
}

// 128x128 transpose+convert: out[n*128+k] = in[k*128+n]
__global__ void k_t128(const float* __restrict__ in, _Float16* __restrict__ out) {
    int i = blockIdx.x * blockDim.x + threadIdx.x;     // 16384
    int k = i >> 7, n = i & 127;
    out[(size_t)n * HDIM + k] = (_Float16)in[i];
}

// h = emb[x-1] in both f32 and f16
__global__ void k_gather(const int* __restrict__ x,
                         const float* __restrict__ emb,
                         const _Float16* __restrict__ emb16,
                         float* __restrict__ h32, _Float16* __restrict__ h16,
                         int total) {
    int i = blockIdx.x * blockDim.x + threadIdx.x;
    if (i >= total) return;
    int row = i >> 7, c = i & 127;
    size_t src = (size_t)(x[row] - 1) * HDIM + c;
    h32[i] = emb[src];
    h16[i] = emb16[src];
}

// edge scatter-add: agg[dst] += m[src]  (native global_atomic_add_f32)
__global__ void k_scatter(const int* __restrict__ ei,
                          const float* __restrict__ m,
                          float* __restrict__ agg, int ne) {
    int i = blockIdx.x * blockDim.x + threadIdx.x;
    if (i >= ne * HDIM) return;
    int e = i >> 7, c = i & 127;
    int src = ei[e];
    int dst = ei[ne + e];
    unsafeAtomicAdd(&agg[(size_t)dst * HDIM + c], m[(size_t)src * HDIM + c]);
}

// GRUCell + relu; gi/gh carry biases already (GEMM epilogue)
__global__ void k_gru(const _Float16* __restrict__ gi,
                      const _Float16* __restrict__ gh,
                      const float* __restrict__ h,
                      float* __restrict__ h2, _Float16* __restrict__ h2h,
                      int total) {
    int i = blockIdx.x * blockDim.x + threadIdx.x;
    if (i >= total) return;
    int row = i >> 7, c = i & 127;
    size_t b3 = (size_t)row * (3 * HDIM);
    float ir = (float)gi[b3 + c], iz = (float)gi[b3 + HDIM + c], in_ = (float)gi[b3 + 2 * HDIM + c];
    float hr = (float)gh[b3 + c], hz = (float)gh[b3 + HDIM + c], hn  = (float)gh[b3 + 2 * HDIM + c];
    float r = fast_sigmoid(ir + hr);
    float z = fast_sigmoid(iz + hz);
    float ng = fast_tanh(in_ + r * hn);
    float hv = (1.0f - z) * ng + z * h[i];
    hv = fmaxf(hv, 0.0f);
    h2[i] = hv;
    h2h[i] = (_Float16)hv;
}

// alpha[i] = a[i,:] . q_w + q_b  (one row per wave, shuffle reduce)
__global__ void k_alpha(const float* __restrict__ a, const float* __restrict__ qw,
                        const float* __restrict__ qb, float* __restrict__ alpha,
                        int nrows) {
    int lane = threadIdx.x & 31;
    int wave = blockIdx.x * (blockDim.x >> 5) + (threadIdx.x >> 5);
    if (wave >= nrows) return;
    const float* row = a + (size_t)wave * HDIM;
    float s = 0.0f;
#pragma unroll
    for (int j = 0; j < HDIM / 32; ++j) s += row[lane + j * 32] * qw[lane + j * 32];
#pragma unroll
    for (int off = 16; off; off >>= 1) s += __shfl_down(s, off, 32);
    if (lane == 0) alpha[wave] = s + qb[0];
}

// per-session: s_g = sum alpha*h2 ; concat16 = [v_n | s_g] as f16 [SESS,256]
__global__ void k_readout(const float* __restrict__ h2,
                          const float* __restrict__ alpha,
                          _Float16* __restrict__ cc) {
    int b = blockIdx.x;        // SESS blocks
    int c = threadIdx.x;       // HDIM threads
    float s = 0.0f;
    for (int j = 0; j < NPER; ++j) {
        int row = b * NPER + j;
        s += alpha[row] * h2[(size_t)row * HDIM + c];
    }
    float vn = h2[(size_t)(b * NPER + NPER - 1) * HDIM + c];
    cc[(size_t)b * 256 + c]        = (_Float16)vn;
    cc[(size_t)b * 256 + HDIM + c] = (_Float16)s;
}

// ---------------------------------------------------------------------------
extern "C" void kernel_launch(void* const* d_in, const int* in_sizes, int n_in,
                              void* d_out, int out_size, void* d_ws, size_t ws_size,
                              hipStream_t stream) {
    (void)in_sizes; (void)n_in; (void)out_size; (void)ws_size;
    const int*   x   = (const int*)d_in[0];
    const int*   ei  = (const int*)d_in[1];
    const float* emb = (const float*)d_in[3];
    const float* Wg  = (const float*)d_in[4];
    const float* Wih = (const float*)d_in[5];
    const float* Whh = (const float*)d_in[6];
    const float* bih = (const float*)d_in[7];
    const float* bhh = (const float*)d_in[8];
    const float* W1w = (const float*)d_in[9];
    const float* W1b = (const float*)d_in[10];
    const float* W2w = (const float*)d_in[11];
    const float* W2b = (const float*)d_in[12];
    const float* qw  = (const float*)d_in[13];
    const float* qb  = (const float*)d_in[14];
    const float* W3w = (const float*)d_in[15];
    const float* W3b = (const float*)d_in[16];
    float* out = (float*)d_out;

    char* ws = (char*)d_ws;
    size_t off = 0;
    auto alloc = [&](size_t bytes) {
        size_t o = off; off += (bytes + 255) & ~(size_t)255; return o;
    };
    _Float16* EMB16 = (_Float16*)(ws + alloc((size_t)NNODE * HDIM * 2));
    _Float16* H16   = (_Float16*)(ws + alloc((size_t)NROWS * HDIM * 2));
    float*    H32   = (float*)   (ws + alloc((size_t)NROWS * HDIM * 4));
    float*    M32   = (float*)   (ws + alloc((size_t)NROWS * HDIM * 4));
    float*    AGG32 = (float*)   (ws + alloc((size_t)NROWS * HDIM * 4));
    _Float16* AGG16 = (_Float16*)(ws + alloc((size_t)NROWS * HDIM * 2));
    _Float16* GI    = (_Float16*)(ws + alloc((size_t)NROWS * 3 * HDIM * 2));
    _Float16* GH    = (_Float16*)(ws + alloc((size_t)NROWS * 3 * HDIM * 2));
    _Float16* WGT16 = (_Float16*)(ws + alloc(128 * 128 * 2));
    _Float16* WIH16 = (_Float16*)(ws + alloc(384 * 128 * 2));
    _Float16* WHH16 = (_Float16*)(ws + alloc(384 * 128 * 2));
    _Float16* W1_16 = (_Float16*)(ws + alloc(128 * 128 * 2));
    _Float16* W2_16 = (_Float16*)(ws + alloc(128 * 128 * 2));
    _Float16* W3_16 = (_Float16*)(ws + alloc(128 * 256 * 2));
    float*    ALPHA = (float*)   (ws + alloc((size_t)NROWS * 4));
    _Float16* CC16  = (_Float16*)(ws + alloc((size_t)SESS * 256 * 2));
    _Float16* SH16  = (_Float16*)(ws + alloc((size_t)SESS * HDIM * 2));
    // buffer reuse (lifetimes disjoint):
    float*    H2_32 = AGG32;   // post-GRU hidden (agg dead after GI gemm)
    _Float16* H2_16 = H16;     // post-GRU f16 (pre-GRU h16 dead after gh/m gemms)
    float*    A32   = M32;     // attention activations (m dead after scatter)

    const int BS = 256;
    auto blks = [](long t, int bs) { return (int)((t + bs - 1) / bs); };
    auto gemm_blks = [](int M, int Nout, int mrep) {
        long tiles = (long)(M / (16 * mrep)) * (Nout / 16);
        return (int)((tiles + 7) / 8);           // 8 waves / 256-thread block
    };

    // ---- weight & embedding prep (f16, B laid out as [Nout,K]) ----
    k_f32_to_f16<<<blks((long)NNODE * HDIM, BS), BS, 0, stream>>>(emb, EMB16, NNODE * HDIM);
    k_f32_to_f16<<<blks(384 * 128, BS), BS, 0, stream>>>(Wih, WIH16, 384 * 128);
    k_f32_to_f16<<<blks(384 * 128, BS), BS, 0, stream>>>(Whh, WHH16, 384 * 128);
    k_f32_to_f16<<<blks(128 * 128, BS), BS, 0, stream>>>(W1w, W1_16, 128 * 128);
    k_f32_to_f16<<<blks(128 * 128, BS), BS, 0, stream>>>(W2w, W2_16, 128 * 128);
    k_f32_to_f16<<<blks(128 * 256, BS), BS, 0, stream>>>(W3w, W3_16, 128 * 256);
    k_t128<<<blks(128 * 128, BS), BS, 0, stream>>>(Wg, WGT16);   // used as B => transpose

    // ---- h = emb[x-1] ----
    k_gather<<<blks((long)NROWS * HDIM, BS), BS, 0, stream>>>(x, emb, EMB16, H32, H16, NROWS * HDIM);

    // ---- m = h @ W_gconv ----
    k_gemm<1, 4, false, 0, true, false><<<gemm_blks(NROWS, HDIM, 1), BS, 0, stream>>>(
        H16, HDIM, WGT16, nullptr, HDIM, nullptr, nullptr,
        M32, nullptr, HDIM, NROWS, HDIM);

    // ---- agg = scatter_add(m[src] -> dst) ----
    k_zero_f32<<<blks((long)NROWS * HDIM, BS), BS, 0, stream>>>(AGG32, NROWS * HDIM);
    k_scatter<<<blks((long)NEDGE * HDIM, BS), BS, 0, stream>>>(ei, M32, AGG32, NEDGE);
    k_f32_to_f16<<<blks((long)NROWS * HDIM, BS), BS, 0, stream>>>(AGG32, AGG16, NROWS * HDIM);

    // ---- gi = agg @ W_ih.T + b_ih ; gh = h @ W_hh.T + b_hh ----
    k_gemm<1, 4, false, 0, false, true><<<gemm_blks(NROWS, 384, 1), BS, 0, stream>>>(
        AGG16, HDIM, WIH16, nullptr, HDIM, bih, nullptr,
        nullptr, GI, 384, NROWS, 384);
    k_gemm<1, 4, false, 0, false, true><<<gemm_blks(NROWS, 384, 1), BS, 0, stream>>>(
        H16, HDIM, WHH16, nullptr, HDIM, bhh, nullptr,
        nullptr, GH, 384, NROWS, 384);

    // ---- GRU + relu ----
    k_gru<<<blks((long)NROWS * HDIM, BS), BS, 0, stream>>>(GI, GH, H32, H2_32, H2_16, NROWS * HDIM);

    // ---- a = sigmoid(v_n[batch]@W1.T + W1_b + h@W2.T + W2_b)  (fused dual-A) ----
    k_gemm<1, 4, true, 1, true, false><<<gemm_blks(NROWS, HDIM, 1), BS, 0, stream>>>(
        H2_16, HDIM, W2_16, W1_16, HDIM, W1b, W2b,
        A32, nullptr, HDIM, NROWS, HDIM);

    // ---- alpha = a @ q_w.T + q_b ; readout into [v_n | s_g] ----
    k_alpha<<<blks((long)NROWS * 32, BS), BS, 0, stream>>>(A32, qw, qb, ALPHA, NROWS);
    k_readout<<<SESS, HDIM, 0, stream>>>(H2_32, ALPHA, CC16);

    // ---- s_h = concat @ W3.T + W3_b ----
    k_gemm<1, 8, false, 0, false, true><<<gemm_blks(SESS, HDIM, 1), BS, 0, stream>>>(
        CC16, 256, W3_16, nullptr, 256, W3b, nullptr,
        nullptr, SH16, HDIM, SESS, HDIM);

    // ---- scores = s_h @ emb.T  (MREP=4: reuse B frags across 4 M-tiles) ----
    k_gemm<4, 4, false, 0, true, false><<<gemm_blks(SESS, NNODE, 4), BS, 0, stream>>>(
        SH16, HDIM, EMB16, nullptr, HDIM, nullptr, nullptr,
        out, nullptr, NNODE, SESS, NNODE);
}
